// FlexBertGLUMoE_53077205843993
// MI455X (gfx1250) — matrix operand order repaired and checked by reading, not compiled, not run on hardware
//
#include <hip/hip_runtime.h>
#include <math.h>

// ---------------------------------------------------------------------------
// Problem constants (from reference): B=2, C=2048, D=1024, I=2048, E=8, K=2
// ---------------------------------------------------------------------------
#define DIM   1024
#define IDIM  2048
#define TWOI  4096
#define NEXP  8
#define NTOK  4096          // B*C
#define NSLOT 8192          // NTOK * K
#define CAP   1280          // int(NTOK * K * 1.25 / E)

// GEMM tiling
#define BM 128
#define BN 64
#define BK 32
#define LDSTR 48            // padded LDS row stride (bf16 elems) -> 96B, 16B aligned
                            // matches TDM pad: 64B row + 32B pad

typedef __attribute__((ext_vector_type(16))) __bf16 v16bf;
typedef __attribute__((ext_vector_type(8)))  float   v8f;
typedef __attribute__((ext_vector_type(8)))  __bf16  bf16x8;
typedef __attribute__((ext_vector_type(4)))  float   f32x4;
typedef __attribute__((ext_vector_type(4)))  unsigned int v4u;
typedef __attribute__((ext_vector_type(8)))  int     v8i;
typedef __attribute__((ext_vector_type(4)))  int     v4i;

// ---------------------------------------------------------------------------
// Tensor Data Mover: load one 2-D bf16 tile (tile_d0 x tile_d1 elements) from
// global (row pitch stride_d0 elements) into LDS at lds_addr, inserting 8
// DWORDs of padding after every 16 DWORDs (64B row -> 96B LDS pitch = LDSTR).
// D# layout per CDNA5 ISA section 8.3/8.4. Caller gates to one wave per
// workgroup via a uniform branch; completion via s_wait_tensorcnt.
// 6-arg clang-23 builtin form: (g0, g1, g2, g3, g4, cpol).
// ---------------------------------------------------------------------------
__device__ __forceinline__ void tdm_load_tile_2d(unsigned lds_addr,
                                                 unsigned long long gaddr,
                                                 unsigned tensor_d0, unsigned tensor_d1,
                                                 unsigned stride_d0,
                                                 unsigned tile_d0, unsigned tile_d1) {
  v4u g0;
  g0[0] = 1u;                                         // count=1, user descriptor
  g0[1] = lds_addr;                                   // lds_addr [63:32]
  g0[2] = (unsigned)gaddr;                            // global_addr lo
  g0[3] = (unsigned)(gaddr >> 32) | (2u << 30);       // global_addr hi | type=2
  v8i g1;
  // workgroup_mask=0 | data_size=1(2B)<<16 | pad_enable<<20 |
  // pad_interval=3(16 DW)<<22 | pad_amount=7(8 DW)<<25
  g1[0] = (int)((1u << 16) | (1u << 20) | (3u << 22) | (7u << 25));
  g1[1] = (int)((tensor_d0 & 0xFFFFu) << 16);                         // tensor_dim0[15:0]
  g1[2] = (int)((tensor_d0 >> 16) | ((tensor_d1 & 0xFFFFu) << 16));   // td0 hi | td1 lo
  g1[3] = (int)((tensor_d1 >> 16) | (tile_d0 << 16));                 // td1 hi | tile_dim0
  g1[4] = (int)(tile_d1 & 0xFFFFu);                                   // tile_dim1 | tile_dim2=0
  g1[5] = (int)stride_d0;                                             // dim0_stride[31:0]
  g1[6] = 0;                                                          // stride hi | dim1_stride lo
  g1[7] = 0;
  const v4i z4 = {0, 0, 0, 0};                        // 2-D tensor: groups 2/3 unused
  const v8i z8 = {0, 0, 0, 0, 0, 0, 0, 0};
  __builtin_amdgcn_tensor_load_to_lds(g0, g1, z4, z4, z8, 0);
}

// ---------------------------------------------------------------------------
// Fragment loaders (per CDNA5 ISA 16-bit A / B VGPR layouts, wave32)
// ---------------------------------------------------------------------------
__device__ __forceinline__ v16bf load_fragA(const __bf16* base, int hl) {
  union { v16bf v; bf16x8 h[2]; } u;
  u.h[0] = *reinterpret_cast<const bf16x8*>(base + hl * 8);
  u.h[1] = *reinterpret_cast<const bf16x8*>(base + 16 + hl * 8);
  return u.v;
}
__device__ __forceinline__ v16bf load_fragB(const __bf16* base, int hl) {
  union { v16bf v; bf16x8 h[2]; } u;
  u.h[0] = *reinterpret_cast<const bf16x8*>(base + hl * 8);
  u.h[1] = *reinterpret_cast<const bf16x8*>(base + 16 + hl * 8);
  return u.v;
}
__device__ __forceinline__ float gelu_exact(float v) {
  return 0.5f * v * (1.0f + erff(v * 0.70710678118654752440f));
}

// ---------------------------------------------------------------------------
// 1) Router: logits = x @ Wg^T, top-2 + softmax. One wave32 per token.
// ---------------------------------------------------------------------------
__global__ __launch_bounds__(256)
void moe_router_kernel(const float* __restrict__ x, const float* __restrict__ Wg,
                       int* __restrict__ tok_e, float* __restrict__ tok_g) {
  const int wave  = threadIdx.x >> 5;
  const int lane  = threadIdx.x & 31;
  const int token = blockIdx.x * 8 + wave;
  if (token >= NTOK) return;
  const float* xr = x + (size_t)token * DIM;
  float acc[NEXP];
#pragma unroll
  for (int e = 0; e < NEXP; ++e) acc[e] = 0.f;
  for (int t = lane; t < DIM; t += 32) {
    const float xv = xr[t];
#pragma unroll
    for (int e = 0; e < NEXP; ++e) acc[e] += xv * Wg[e * DIM + t];
  }
#pragma unroll
  for (int off = 16; off > 0; off >>= 1) {
#pragma unroll
    for (int e = 0; e < NEXP; ++e) acc[e] += __shfl_xor(acc[e], off, 32);
  }
  if (lane == 0) {
    float v0 = -3.4e38f, v1 = -3.4e38f;
    int   i0 = 0, i1 = 0;
#pragma unroll
    for (int e = 0; e < NEXP; ++e) {
      const float v = acc[e];
      if (v > v0)      { v1 = v0; i1 = i0; v0 = v; i0 = e; }
      else if (v > v1) { v1 = v;  i1 = e; }
    }
    const float g0 = 1.f / (1.f + expf(v1 - v0));   // softmax over top-2
    tok_e[2 * token]     = i0;
    tok_e[2 * token + 1] = i1;
    tok_g[2 * token]     = g0;
    tok_g[2 * token + 1] = 1.f - g0;
  }
}

// ---------------------------------------------------------------------------
// 2) Init slot_token = -1 (workspace is poisoned between calls)
// ---------------------------------------------------------------------------
__global__ void moe_fill_kernel(int* __restrict__ slot_token) {
  const int i = blockIdx.x * blockDim.x + threadIdx.x;
  if (i < NEXP * CAP) slot_token[i] = -1;
}

// ---------------------------------------------------------------------------
// 3) Deterministic greedy capacity assignment: blocked prefix scan over the
//    8192 slots in reference order (token-major, then k).
// ---------------------------------------------------------------------------
__global__ __launch_bounds__(256)
void moe_assign_kernel(const int* __restrict__ tok_e, int* __restrict__ tok_pos,
                       int* __restrict__ slot_token) {
  __shared__ int cnts[256][NEXP];
  const int t   = threadIdx.x;
  const int per = NSLOT / 256;    // 32 slots per thread
  int local[NEXP];
#pragma unroll
  for (int e = 0; e < NEXP; ++e) local[e] = 0;
  const int s0 = t * per;
  for (int s = s0; s < s0 + per; ++s) local[tok_e[s]]++;
#pragma unroll
  for (int e = 0; e < NEXP; ++e) cnts[t][e] = local[e];
  __syncthreads();
  if (t < NEXP) {                 // exclusive prefix per expert over thread chunks
    int run = 0;
    for (int i = 0; i < 256; ++i) { const int c = cnts[i][t]; cnts[i][t] = run; run += c; }
  }
  __syncthreads();
  int offs[NEXP];
#pragma unroll
  for (int e = 0; e < NEXP; ++e) offs[e] = cnts[t][e];
  for (int s = s0; s < s0 + per; ++s) {
    const int e   = tok_e[s];
    const int pos = offs[e]++;
    if (pos < CAP) { tok_pos[s] = pos; slot_token[e * CAP + pos] = s >> 1; }
    else           { tok_pos[s] = -1; }
  }
}

// ---------------------------------------------------------------------------
// 4) Dispatch: exp_x[e, pos, :] = bf16(x[token]); zero empty capacity rows.
// ---------------------------------------------------------------------------
__global__ __launch_bounds__(256)
void moe_gather_kernel(const float* __restrict__ x, const int* __restrict__ slot_token,
                       __bf16* __restrict__ exp_x) {
  const int slot = blockIdx.x;              // e*CAP + pos
  const int tok  = slot_token[slot];
  __bf16* dst = exp_x + (size_t)slot * DIM;
  if (tok < 0) {
    for (int i = threadIdx.x; i < DIM; i += 256) dst[i] = (__bf16)0.f;
  } else {
    const float* src = x + (size_t)tok * DIM;
    for (int i = threadIdx.x; i < DIM; i += 256) dst[i] = (__bf16)src[i];
  }
}

// ---------------------------------------------------------------------------
// 5) Expert GEMM1 + GLU:  H[e] = gelu(Xe@W1[:, :I]) * (Xe@W1[:, I:])
//    bf16 WMMA 16x16x32, f32 accumulate. Double-buffered pipeline, one
//    barrier per K-step: TDM A-tile DMA and B-tile global loads for step i+1
//    overlap the WMMA compute of step i.
// ---------------------------------------------------------------------------
__global__ __launch_bounds__(256)
void moe_gemm_glu_kernel(const __bf16* __restrict__ exp_x, const float* __restrict__ W1,
                         __bf16* __restrict__ Hbuf) {
  __shared__ __bf16 As[2][BM * LDSTR];       // 2 x 12 KB (TDM-padded 96B pitch)
  __shared__ __bf16 Bs[2][2][BN * LDSTR];    // 2 stages x (inp/gate) x 6 KB

  const int e  = blockIdx.z;
  const int m0 = blockIdx.y * BM;
  const int n0 = blockIdx.x * BN;
  const __bf16* A  = exp_x + ((size_t)e * CAP + m0) * DIM;
  const float*  Bg = W1 + (size_t)e * DIM * TWOI;

  const int tid  = threadIdx.x;
  const int wave = tid >> 5;
  const int lane = tid & 31;
  const int wm   = (wave & 3) * 32;
  const int wn   = (wave >> 2) * 32;
  const int hl   = lane >> 4;
  const int r    = lane & 15;
  // uniform (SGPR) wave-0 predicate so TDM issue is a scalar branch
  const bool tdm_wave = ((__builtin_amdgcn_readfirstlane(tid) >> 5) == 0);

  const v8f vzero = {0.f, 0.f, 0.f, 0.f, 0.f, 0.f, 0.f, 0.f};
  v8f acc[2][2][2];                          // [glu_half][mt][nt]
#pragma unroll
  for (int g = 0; g < 2; ++g)
#pragma unroll
    for (int mt = 0; mt < 2; ++mt)
#pragma unroll
      for (int nt = 0; nt < 2; ++nt) acc[g][mt][nt] = vzero;

  const int bk   = tid >> 3;                 // B stage: row k, 8 cols x f32x4 x2
  const int bcol = (tid & 7) * 8;

  f32x4 br0[2], br1[2];                      // in-flight B registers (per GLU half)

  // ---- prologue: stage tile 0 into buffer 0 ----
  if (tdm_wave)
    tdm_load_tile_2d((unsigned)(unsigned long long)(uintptr_t)&As[0][0],
                     (unsigned long long)(uintptr_t)A,
                     DIM, CAP, DIM, BK, BM);
#pragma unroll
  for (int g = 0; g < 2; ++g) {
    const float* src = Bg + (size_t)bk * TWOI + n0 + g * IDIM + bcol;
    br0[g] = *reinterpret_cast<const f32x4*>(src);
    br1[g] = *reinterpret_cast<const f32x4*>(src + 4);
  }
#pragma unroll
  for (int g = 0; g < 2; ++g) {
#pragma unroll
    for (int i = 0; i < 4; ++i) Bs[0][g][(bcol + i)     * LDSTR + bk] = (__bf16)br0[g][i];
#pragma unroll
    for (int i = 0; i < 4; ++i) Bs[0][g][(bcol + 4 + i) * LDSTR + bk] = (__bf16)br1[g][i];
  }
  if (tdm_wave) __builtin_amdgcn_s_wait_tensorcnt(0);
  __syncthreads();

  // ---- pipelined main loop: one barrier per K-step ----
  const int NK = DIM / BK;
  for (int it = 0; it < NK; ++it) {
    const int cbuf = it & 1;
    const int nbuf = cbuf ^ 1;
    const int kn   = (it + 1) * BK;
    const bool have_next = (kn < DIM);

    if (have_next) {
      if (tdm_wave)                          // async DMA of next A tile
        tdm_load_tile_2d((unsigned)(unsigned long long)(uintptr_t)&As[nbuf][0],
                         (unsigned long long)(uintptr_t)(A + kn),
                         DIM, CAP, DIM, BK, BM);
#pragma unroll
      for (int g = 0; g < 2; ++g) {          // next B tile -> registers (overlaps WMMA)
        const float* src = Bg + (size_t)(kn + bk) * TWOI + n0 + g * IDIM + bcol;
        br0[g] = *reinterpret_cast<const f32x4*>(src);
        br1[g] = *reinterpret_cast<const f32x4*>(src + 4);
      }
      if (kn + BK < DIM)                     // warm L2 for step i+2
        __builtin_prefetch(Bg + (size_t)(kn + BK + bk) * TWOI + n0 + bcol, 0, 1);
    }

    // compute on current buffers
    v16bf af[2];
    af[0] = load_fragA(&As[cbuf][(wm + r) * LDSTR], hl);
    af[1] = load_fragA(&As[cbuf][(wm + 16 + r) * LDSTR], hl);
#pragma unroll
    for (int g = 0; g < 2; ++g) {
#pragma unroll
      for (int nt = 0; nt < 2; ++nt) {
        const v16bf bf = load_fragB(&Bs[cbuf][g][(wn + nt * 16 + r) * LDSTR], hl);
#pragma unroll
        for (int mt = 0; mt < 2; ++mt) {
          acc[g][mt][nt] = __builtin_amdgcn_wmma_f32_16x16x32_bf16(
              false, af[mt], false, bf, (short)0, acc[g][mt][nt], false, false);
        }
      }
    }

    if (have_next) {                         // publish next B tile, drain TDM
#pragma unroll
      for (int g = 0; g < 2; ++g) {
#pragma unroll
        for (int i = 0; i < 4; ++i) Bs[nbuf][g][(bcol + i)     * LDSTR + bk] = (__bf16)br0[g][i];
#pragma unroll
        for (int i = 0; i < 4; ++i) Bs[nbuf][g][(bcol + 4 + i) * LDSTR + bk] = (__bf16)br1[g][i];
      }
      if (tdm_wave) __builtin_amdgcn_s_wait_tensorcnt(0);
    }
    __syncthreads();
  }

  // Epilogue: GLU, store bf16 H.  C layout: VGPR i -> row 8*hl + i, col r.
  __bf16* Ho = Hbuf + ((size_t)e * CAP + m0) * IDIM;
#pragma unroll
  for (int mt = 0; mt < 2; ++mt)
#pragma unroll
    for (int nt = 0; nt < 2; ++nt) {
      const int col = n0 + wn + nt * 16 + r;
#pragma unroll
      for (int i = 0; i < 8; ++i) {
        const float h   = gelu_exact(acc[0][mt][nt][i]) * acc[1][mt][nt][i];
        const int   row = wm + mt * 16 + hl * 8 + i;
        Ho[(size_t)row * IDIM + col] = (__bf16)h;
      }
    }
}

// ---------------------------------------------------------------------------
// 6) Expert GEMM2: exp_out[e] = H[e] @ W2[e]   (f32 output), same pipeline
// ---------------------------------------------------------------------------
__global__ __launch_bounds__(256)
void moe_gemm2_kernel(const __bf16* __restrict__ Hbuf, const float* __restrict__ W2,
                      float* __restrict__ exp_out) {
  __shared__ __bf16 As[2][BM * LDSTR];
  __shared__ __bf16 Bs[2][BN * LDSTR];

  const int e  = blockIdx.z;
  const int m0 = blockIdx.y * BM;
  const int n0 = blockIdx.x * BN;
  const __bf16* A  = Hbuf + ((size_t)e * CAP + m0) * IDIM;
  const float*  Bg = W2 + (size_t)e * IDIM * DIM;

  const int tid  = threadIdx.x;
  const int wave = tid >> 5;
  const int lane = tid & 31;
  const int wm   = (wave & 3) * 32;
  const int wn   = (wave >> 2) * 32;
  const int hl   = lane >> 4;
  const int r    = lane & 15;
  const bool tdm_wave = ((__builtin_amdgcn_readfirstlane(tid) >> 5) == 0);

  const v8f vzero = {0.f, 0.f, 0.f, 0.f, 0.f, 0.f, 0.f, 0.f};
  v8f acc[2][2];
#pragma unroll
  for (int mt = 0; mt < 2; ++mt)
#pragma unroll
    for (int nt = 0; nt < 2; ++nt) acc[mt][nt] = vzero;

  const int bk   = tid >> 3;
  const int bcol = (tid & 7) * 8;

  f32x4 br0, br1;

  // ---- prologue ----
  if (tdm_wave)
    tdm_load_tile_2d((unsigned)(unsigned long long)(uintptr_t)&As[0][0],
                     (unsigned long long)(uintptr_t)A,
                     IDIM, CAP, IDIM, BK, BM);
  {
    const float* src = Bg + (size_t)bk * DIM + n0 + bcol;
    br0 = *reinterpret_cast<const f32x4*>(src);
    br1 = *reinterpret_cast<const f32x4*>(src + 4);
  }
#pragma unroll
  for (int i = 0; i < 4; ++i) Bs[0][(bcol + i)     * LDSTR + bk] = (__bf16)br0[i];
#pragma unroll
  for (int i = 0; i < 4; ++i) Bs[0][(bcol + 4 + i) * LDSTR + bk] = (__bf16)br1[i];
  if (tdm_wave) __builtin_amdgcn_s_wait_tensorcnt(0);
  __syncthreads();

  // ---- pipelined main loop ----
  const int NK = IDIM / BK;
  for (int it = 0; it < NK; ++it) {
    const int cbuf = it & 1;
    const int nbuf = cbuf ^ 1;
    const int kn   = (it + 1) * BK;
    const bool have_next = (kn < IDIM);

    if (have_next) {
      if (tdm_wave)
        tdm_load_tile_2d((unsigned)(unsigned long long)(uintptr_t)&As[nbuf][0],
                         (unsigned long long)(uintptr_t)(A + kn),
                         IDIM, CAP, IDIM, BK, BM);
      const float* src = Bg + (size_t)(kn + bk) * DIM + n0 + bcol;
      br0 = *reinterpret_cast<const f32x4*>(src);
      br1 = *reinterpret_cast<const f32x4*>(src + 4);
      if (kn + BK < IDIM)
        __builtin_prefetch(Bg + (size_t)(kn + BK + bk) * DIM + n0 + bcol, 0, 1);
    }

    v16bf af[2];
    af[0] = load_fragA(&As[cbuf][(wm + r) * LDSTR], hl);
    af[1] = load_fragA(&As[cbuf][(wm + 16 + r) * LDSTR], hl);
#pragma unroll
    for (int nt = 0; nt < 2; ++nt) {
      const v16bf bf = load_fragB(&Bs[cbuf][(wn + nt * 16 + r) * LDSTR], hl);
#pragma unroll
      for (int mt = 0; mt < 2; ++mt) {
        acc[mt][nt] = __builtin_amdgcn_wmma_f32_16x16x32_bf16(
            false, af[mt], false, bf, (short)0, acc[mt][nt], false, false);
      }
    }

    if (have_next) {
#pragma unroll
      for (int i = 0; i < 4; ++i) Bs[nbuf][(bcol + i)     * LDSTR + bk] = (__bf16)br0[i];
#pragma unroll
      for (int i = 0; i < 4; ++i) Bs[nbuf][(bcol + 4 + i) * LDSTR + bk] = (__bf16)br1[i];
      if (tdm_wave) __builtin_amdgcn_s_wait_tensorcnt(0);
    }
    __syncthreads();
  }

  float* O = exp_out + ((size_t)e * CAP + m0) * DIM;
#pragma unroll
  for (int mt = 0; mt < 2; ++mt)
#pragma unroll
    for (int nt = 0; nt < 2; ++nt) {
      const int col = n0 + wn + nt * 16 + r;
#pragma unroll
      for (int i = 0; i < 8; ++i) {
        const int row = wm + mt * 16 + hl * 8 + i;
        O[(size_t)row * DIM + col] = acc[mt][nt][i];
      }
    }
}

// ---------------------------------------------------------------------------
// 7) Combine: out[n] = sum_k kept ? gate_k * exp_out[e_k, pos_k] : 0
// ---------------------------------------------------------------------------
__global__ __launch_bounds__(256)
void moe_combine_kernel(const int* __restrict__ tok_e, const int* __restrict__ tok_pos,
                        const float* __restrict__ tok_g, const float* __restrict__ exp_out,
                        float* __restrict__ out) {
  const int n  = blockIdx.x;
  const int e0 = tok_e[2 * n],   e1 = tok_e[2 * n + 1];
  const int p0 = tok_pos[2 * n], p1 = tok_pos[2 * n + 1];
  const float g0 = tok_g[2 * n], g1 = tok_g[2 * n + 1];
  const float* r0 = exp_out + ((size_t)e0 * CAP + (p0 < 0 ? 0 : p0)) * DIM;
  const float* r1 = exp_out + ((size_t)e1 * CAP + (p1 < 0 ? 0 : p1)) * DIM;
  for (int i = threadIdx.x; i < DIM; i += 256) {
    float v = 0.f;
    if (p0 >= 0) v += g0 * r0[i];
    if (p1 >= 0) v += g1 * r1[i];
    out[(size_t)n * DIM + i] = v;
  }
}

// ---------------------------------------------------------------------------
// Launch
// ---------------------------------------------------------------------------
extern "C" void kernel_launch(void* const* d_in, const int* in_sizes, int n_in,
                              void* d_out, int out_size, void* d_ws, size_t ws_size,
                              hipStream_t stream) {
  (void)in_sizes; (void)n_in; (void)out_size; (void)ws_size;
  const float* x  = (const float*)d_in[0];   // [2,2048,1024]
  const float* Wg = (const float*)d_in[1];   // [8,1024]
  const float* W1 = (const float*)d_in[2];   // [8,1024,4096]
  const float* W2 = (const float*)d_in[3];   // [8,2048,1024]
  float* out = (float*)d_out;                // [2,2048,1024] f32

  char* ws = (char*)d_ws;
  size_t off = 0;
  auto alloc = [&](size_t bytes) -> void* {
    off = (off + 255) & ~(size_t)255;
    void* p = ws + off;
    off += bytes;
    return p;
  };
  int*    tok_e      = (int*)   alloc((size_t)NSLOT * sizeof(int));
  int*    tok_pos    = (int*)   alloc((size_t)NSLOT * sizeof(int));
  float*  tok_g      = (float*) alloc((size_t)NSLOT * sizeof(float));
  int*    slot_token = (int*)   alloc((size_t)NEXP * CAP * sizeof(int));
  __bf16* exp_x      = (__bf16*)alloc((size_t)NEXP * CAP * DIM * sizeof(__bf16));
  __bf16* Hbuf       = (__bf16*)alloc((size_t)NEXP * CAP * IDIM * sizeof(__bf16));
  float*  exp_out    = (float*) alloc((size_t)NEXP * CAP * DIM * sizeof(float));

  moe_router_kernel<<<NTOK / 8, 256, 0, stream>>>(x, Wg, tok_e, tok_g);
  moe_fill_kernel<<<(NEXP * CAP + 255) / 256, 256, 0, stream>>>(slot_token);
  moe_assign_kernel<<<1, 256, 0, stream>>>(tok_e, tok_pos, slot_token);
  moe_gather_kernel<<<NEXP * CAP, 256, 0, stream>>>(x, slot_token, exp_x);

  dim3 g1(IDIM / BN, CAP / BM, NEXP);   // 32 x 10 x 8
  moe_gemm_glu_kernel<<<g1, 256, 0, stream>>>(exp_x, W1, Hbuf);

  dim3 g2(DIM / BN, CAP / BM, NEXP);    // 16 x 10 x 8
  moe_gemm2_kernel<<<g2, 256, 0, stream>>>(Hbuf, W2, exp_out);

  moe_combine_kernel<<<NTOK, 256, 0, stream>>>(tok_e, tok_pos, tok_g, exp_out, out);
}
